// ScaledDotProductAttentionClass_15857019257096
// MI455X (gfx1250) — compile-verified, matrix operand order checked by reading, not run on hardware
//
#include <hip/hip_runtime.h>
#include <hip/hip_bf16.h>

// CDNA5 / gfx1250 attention kernel: bf16 WMMA (v_wmma_f32_16x16x32_bf16),
// two-pass online-softmax + winner-take-all mask.
// Round 2: double-buffered K staging (1 barrier/iter, loads overlap WMMA),
//          sQ/sOut LDS union (49.5 KB total -> 2 workgroups per WGP).
// Round 3: use ext_vector float4 (v4f) so nontemporal builtins accept it.

typedef __attribute__((ext_vector_type(16))) __bf16 v16bf;
typedef __attribute__((ext_vector_type(8)))  float  v8f;
typedef __attribute__((ext_vector_type(4)))  float  v4f;

#define B_   32
#define L_   2048
#define D_   128
#define BQ   64        // q rows per block
#define NT   256       // threads per block (8 wave32)
#define KTP  64        // k-tile pairs (128 k-tiles of 16)

__device__ __forceinline__ void cvt_store4(__bf16* dst, v4f f) {
  union { uint2 u; __bf16 h[4]; } pk;
  pk.h[0] = (__bf16)f.x; pk.h[1] = (__bf16)f.y;
  pk.h[2] = (__bf16)f.z; pk.h[3] = (__bf16)f.w;
  *(uint2*)dst = pk.u;
}

// Load one 16x32 bf16 A/B fragment slice from a row-major LDS tile row.
// Per-lane layout (ISA 7.12.2, 16-bit A; B mirrors with M<->N):
//   elems[0..7]  = d (dbase+off+0 .. +7)
//   elems[8..15] = d (dbase+off+16 .. +23),  off = lane<16 ? 0 : 8
__device__ __forceinline__ v16bf load_frag(const __bf16* rowp, int dbase, int off) {
  union { v16bf v; uint4 q[2]; } f;
  f.q[0] = *(const uint4*)(rowp + dbase + off);
  f.q[1] = *(const uint4*)(rowp + dbase + off + 16);
  return f.v;
}

struct SmemT {
  union {
    __align__(16) __bf16 sQ[BQ][D_];     // 16 KB, live only before pass A
    __align__(16) float  sOut[BQ][D_];   // 32 KB, live only in pass B
  } u;
  __align__(16) __bf16 sK[2][2][16][D_]; // 16 KB: [stage-buf][parity][row][d]
  float sMp[2][BQ];
  float sLp[2][BQ];
  float sM[BQ];
  float sL[BQ];
};

// Stage one k-tile pair (32 K rows) fp32 -> bf16 into LDS buffer. All 256 threads.
__device__ __forceinline__ void stage_k(const float* __restrict__ Kg,
                                        __bf16 (*sKb)[16][D_],
                                        int b, int kp, int tid) {
  const int slot = tid >> 7;           // 0..1 (k-tile parity within the pair)
  const int row  = (tid >> 3) & 15;    // 0..15
  const int ch   = tid & 7;            // 0..7 -> 16 floats each
  const int krow = kp * 32 + slot * 16 + row;
  const v4f* src = (const v4f*)(Kg + ((size_t)b * L_ + krow) * D_ + ch * 16);
  if (kp + 2 < KTP)   // distance-2 prefetch (next-next pair, +64 rows)
    __builtin_prefetch((const char*)src + 64 * D_ * sizeof(float), 0, 0);
  __bf16* dst = &sKb[slot][row][ch * 16];
#pragma unroll
  for (int j = 0; j < 4; ++j) cvt_store4(dst + j * 4, src[j]);
}

__global__ __launch_bounds__(NT)
void attn_wta_kernel(const float* __restrict__ Q, const float* __restrict__ K,
                     const float* __restrict__ V, float* __restrict__ Out,
                     float* __restrict__ Attn) {
  __shared__ SmemT sm;

  const int tid  = threadIdx.x;
  const int lane = tid & 31;
  const int wave = tid >> 5;
  const int qsub = wave & 3;       // q-subtile 0..3 (16 rows each)
  const int kpar = wave >> 2;      // k-tile parity 0/1
  const int b    = blockIdx.x >> 5;            // 32 q-tiles per batch
  const int q0   = (blockIdx.x & 31) * BQ;
  const float invT = 0.08838834764831845f;     // 1/sqrt(128)

  // stage Q tile fp32 -> bf16 (64 x 128)
  {
    const int row = tid >> 2;
    const int cb  = (tid & 3) * 32;
    const v4f* src = (const v4f*)(Q + ((size_t)b * L_ + q0 + row) * D_ + cb);
    __bf16* dst = &sm.u.sQ[row][cb];
#pragma unroll
    for (int j = 0; j < 8; ++j) cvt_store4(dst + j * 4, __builtin_nontemporal_load(src + j));
  }
  __syncthreads();

  // Per-lane fragment geometry
  const int off  = (lane < 16) ? 0 : 8;
  const int mrow = lane & 15;                 // A: M row; B: N col; C: N col
  const int qoff = (lane < 16) ? 0 : 8;       // C-layout: VGPR r -> M=r or r+8

  // A fragments for this wave's q-subtile (persist across both passes)
  v16bf afrag[4];
  {
    const __bf16* qrow = sm.u.sQ[qsub * 16 + mrow];
#pragma unroll
    for (int j = 0; j < 4; ++j) afrag[j] = load_frag(qrow, j * 32, off);
  }
  __syncthreads();   // everyone done reading sQ (aliases sOut)

  float mreg[8], lreg[8];
#pragma unroll
  for (int i = 0; i < 8; ++i) { mreg[i] = -__builtin_inff(); lreg[i] = 0.0f; }

  // ---------------- PASS A: online softmax stats ----------------
  stage_k(K, sm.sK[0], b, 0, tid);
#pragma unroll 2
  for (int kp = 0; kp < KTP; ++kp) {
    __syncthreads();   // staging of kp complete; reads of kp-1 complete
    if (kp + 1 < KTP) stage_k(K, sm.sK[(kp + 1) & 1], b, kp + 1, tid);

    const __bf16* krow = sm.sK[kp & 1][kpar][mrow];
    v16bf bfrag[4];
#pragma unroll
    for (int j = 0; j < 4; ++j) bfrag[j] = load_frag(krow, j * 32, off);

    v8f c = {};
#pragma unroll
    for (int j = 0; j < 4; ++j)
      c = __builtin_amdgcn_wmma_f32_16x16x32_bf16(false, afrag[j], false, bfrag[j],
                                                  (short)0, c, false, false);
#pragma unroll
    for (int i = 0; i < 8; ++i) {
      float s  = c[i] * invT;
      float mn = fmaxf(mreg[i], s);
      lreg[i]  = lreg[i] * __expf(mreg[i] - mn) + __expf(s - mn);
      mreg[i]  = mn;
    }
  }

  // reduce stats across the 16 lanes sharing a q-row
#pragma unroll
  for (int i = 0; i < 8; ++i) {
    float m = mreg[i], l = lreg[i];
#pragma unroll
    for (int sw = 1; sw < 16; sw <<= 1) {
      float mo = __shfl_xor(m, sw, 16);
      float lo = __shfl_xor(l, sw, 16);
      float mn = fmaxf(m, mo);
      l = l * __expf(m - mn) + lo * __expf(mo - mn);
      m = mn;
    }
    if (mrow == 0) {
      const int qloc = qsub * 16 + i + qoff;
      sm.sMp[kpar][qloc] = m;
      sm.sLp[kpar][qloc] = l;
    }
  }
  __syncthreads();
  if (tid < BQ) {
    float ma = sm.sMp[0][tid], mb = sm.sMp[1][tid];
    float M  = fmaxf(ma, mb);
    sm.sM[tid] = M;
    sm.sL[tid] = sm.sLp[0][tid] * __expf(ma - M) + sm.sLp[1][tid] * __expf(mb - M);
  }
  __syncthreads();

  float Mv[8], inv[8];
#pragma unroll
  for (int i = 0; i < 8; ++i) {
    const int qloc = qsub * 16 + i + qoff;
    Mv[i]  = sm.sM[qloc];
    inv[i] = 1.0f / sm.sL[qloc];
  }

  // zero output accumulator (aliases the now-dead sQ)
  for (int i = tid; i < BQ * D_; i += NT) ((float*)sm.u.sOut)[i] = 0.0f;

  // ---------------- PASS B: masked attn write + output ----------------
  stage_k(K, sm.sK[0], b, 0, tid);
#pragma unroll 2
  for (int kp = 0; kp < KTP; ++kp) {
    __syncthreads();
    if (kp + 1 < KTP) stage_k(K, sm.sK[(kp + 1) & 1], b, kp + 1, tid);

    const __bf16* krow = sm.sK[kp & 1][kpar][mrow];
    v16bf bfrag[4];
#pragma unroll
    for (int j = 0; j < 4; ++j) bfrag[j] = load_frag(krow, j * 32, off);

    v8f c = {};
#pragma unroll
    for (int j = 0; j < 4; ++j)
      c = __builtin_amdgcn_wmma_f32_16x16x32_bf16(false, afrag[j], false, bfrag[j],
                                                  (short)0, c, false, false);

    const int kt   = kp * 2 + kpar;
    const int kcol = kt * 16 + mrow;
#pragma unroll
    for (int i = 0; i < 8; ++i) {
      float s   = c[i] * invT;             // bitwise-identical to pass A
      float val = (s == Mv[i]) ? inv[i] : 0.0f;
      const int qloc = qsub * 16 + i + qoff;
      const size_t aidx = ((size_t)b * L_ + (q0 + qloc)) * (size_t)L_ + kcol;
      __builtin_nontemporal_store(val, Attn + aidx);   // 537 MB stream: bypass cache
      if (val != 0.0f) {
        const v4f* vp = (const v4f*)(V + ((size_t)b * L_ + kcol) * D_);
#pragma unroll 4
        for (int j = 0; j < 32; ++j) {
          v4f vv = vp[j];
          atomicAdd(&sm.u.sOut[qloc][j * 4 + 0], val * vv.x);
          atomicAdd(&sm.u.sOut[qloc][j * 4 + 1], val * vv.y);
          atomicAdd(&sm.u.sOut[qloc][j * 4 + 2], val * vv.z);
          atomicAdd(&sm.u.sOut[qloc][j * 4 + 3], val * vv.w);
        }
      }
    }
  }
  __syncthreads();   // all atomicAdds into sOut complete

  // write output tile
  {
    const int row = tid >> 2;
    const int cb  = (tid & 3) * 32;
    v4f* op = (v4f*)(Out + ((size_t)b * L_ + q0 + row) * D_ + cb);
    const v4f* sp = (const v4f*)(&sm.u.sOut[row][cb]);
#pragma unroll
    for (int j = 0; j < 8; ++j) op[j] = sp[j];
  }
}

extern "C" void kernel_launch(void* const* d_in, const int* in_sizes, int n_in,
                              void* d_out, int out_size, void* d_ws, size_t ws_size,
                              hipStream_t stream) {
  const float* q = (const float*)d_in[0];
  const float* k = (const float*)d_in[1];
  const float* v = (const float*)d_in[2];
  float* out  = (float*)d_out;
  float* attn = out + (size_t)B_ * L_ * D_;   // outputs concatenated: (output, attn)

  dim3 grid(B_ * (L_ / BQ));   // 1024 blocks
  dim3 block(NT);              // 256 threads = 8 wave32
  attn_wta_kernel<<<grid, block, 0, stream>>>(q, k, v, out, attn);
}
